// Model_71511205478790
// MI455X (gfx1250) — matrix operand order repaired and checked by reading, not compiled, not run on hardware
//
#include <hip/hip_runtime.h>
#include <cmath>

// ---------------------------------------------------------------------------
// Problem geometry
//   H=400 (pad 416 = 13*32), 4H=1600 gate cols, B=64, T=1024, OUT=121 (pad 128)
// ---------------------------------------------------------------------------
typedef __attribute__((ext_vector_type(16))) __bf16 v16bf;
typedef __attribute__((ext_vector_type(8)))  float  v8f;

union AFrag { unsigned int u[8]; v16bf v; };

constexpr int    Hh      = 400;
constexpr int    KP      = 416;               // padded K (13 * 32)
constexpr int    KS      = 13;                // k-steps of 32
constexpr int    B       = 64;
constexpr int    T       = 1024;
constexpr size_t WHH_EL  = (size_t)100 * KS * 32 * 16;   // ushorts per packed 1600x400 matrix
constexpr size_t WL_EL   = (size_t)8 * 39 * 32 * 16;     // ushorts for packed Wl (1248x128)
constexpr size_t OUT_EL  = (size_t)T * B * KP;           // ushorts per layer output (bf16)

constexpr size_t ALN(size_t x) { return (x + 255) & ~(size_t)255; }
constexpr size_t OFF_FLAGS = 0;
constexpr size_t OFF_BX    = ALN(OFF_FLAGS + (size_t)3 * T * sizeof(int));
constexpr size_t OFF_WHH   = ALN(OFF_BX + (size_t)3 * 1600 * sizeof(float));
constexpr size_t OFF_WIHH  = ALN(OFF_WHH + 3 * WHH_EL * 2);
constexpr size_t OFF_WL    = ALN(OFF_WIHH + 2 * WHH_EL * 2);
constexpr size_t OFF_OUT   = ALN(OFF_WL + WL_EL * 2);

__device__ __forceinline__ unsigned short f2bf(float f) {
    union { float f; unsigned u; } v; v.f = f;
    unsigned r = v.u + 0x7FFFu + ((v.u >> 16) & 1u);   // round-to-nearest-even
    return (unsigned short)(r >> 16);
}
__device__ __forceinline__ float sigf(float x) { return 1.f / (1.f + __expf(-x)); }

// ---------------------------------------------------------------------------
// Prep kernels (run every launch: deterministic, graph-replay safe)
// ---------------------------------------------------------------------------
__global__ void k_init_flags(int* flags) {
    int i = blockIdx.x * blockDim.x + threadIdx.x;
    if (i < 3 * T) flags[i] = 0;
}

__global__ void k_pack_bias(const float* a0, const float* b0, const float* a1,
                            const float* b1, const float* a2, const float* b2, float* bx) {
    int i = blockIdx.x * blockDim.x + threadIdx.x;
    if (i >= 4800) return;
    int l = i / 1600, c = i % 1600;
    const float* A = (l == 0) ? a0 : (l == 1) ? a1 : a2;
    const float* Bp = (l == 0) ? b0 : (l == 1) ? b1 : b2;
    bx[i] = A[c] + Bp[c];
}

// Pack a [1600 x Ksrc] row-major weight into WMMA B-fragment order.
// frag index = ((colTile*13 + ks)*32 + lane)*16 + e ;  col = colTile*16 + lane%16
// k = ks*32 + (lane>=16 ? 8:0) + (e>=8 ? 16:0) + (e&7)   (B layout: table order)
__global__ void k_pack_w(const float* W, unsigned short* dst, int colStride, int kOff) {
    int i = blockIdx.x * blockDim.x + threadIdx.x;
    if (i >= (int)WHH_EL) return;
    int e    = i & 15;
    int lane = (i >> 4) & 31;
    int ks   = (i >> 9) % KS;
    int nt   = i / (KS * 512);
    int col  = nt * 16 + (lane & 15);
    int k    = ks * 32 + ((lane & 16) ? 8 : 0) + ((e & 8) ? 16 : 0) + (e & 7);
    float v  = (k < Hh) ? W[(size_t)col * colStride + kOff + k] : 0.f;
    dst[i] = f2bf(v);
}

// Pack Wl [121 x 1200] into B-fragments over padded K=1248 (3 segments of 416).
__global__ void k_pack_wl(const float* Wl, unsigned short* dst) {
    int i = blockIdx.x * blockDim.x + threadIdx.x;
    if (i >= (int)WL_EL) return;
    int e    = i & 15;
    int lane = (i >> 4) & 31;
    int ks   = (i >> 9) % 39;
    int nt   = i / (39 * 512);
    int n    = nt * 16 + (lane & 15);
    int kk   = ks * 32 + ((lane & 16) ? 8 : 0) + ((e & 8) ? 16 : 0) + (e & 7);
    int seg  = kk / KP, w = kk % KP;
    float v  = (n < 121 && w < Hh) ? Wl[(size_t)n * 1200 + seg * Hh + w] : 0.f;
    dst[i] = f2bf(v);
}

// ---------------------------------------------------------------------------
// Pipelined 3-layer LSTM: grid = 3 blocks (one per layer), 512 threads each.
// h kept in LDS (bf16, K-padded); c in f32 regs, staged h in packed-bf16 regs.
// Layers pipeline with 1-step skew via global flags (release/acquire).
// ---------------------------------------------------------------------------
__global__ void __launch_bounds__(512)
k_lstm(const float* __restrict__ X,
       const float* __restrict__ Wih0, const float* __restrict__ Wih1,
       const float* __restrict__ Wih2,
       const float* __restrict__ bx,
       const unsigned short* __restrict__ whhP,
       const unsigned short* __restrict__ wihhP,
       unsigned short* __restrict__ outs,
       int* __restrict__ flags) {
    const int layer = blockIdx.x;
    const int tid   = threadIdx.x;
    const int lane  = tid & 31;
    const int wave  = tid >> 5;

    __shared__ unsigned short hbuf[B * KP];   // 52 KB: h_t (bf16, zero-padded K)
    __shared__ float          xbuf[B * 3];

    for (int i = tid; i < B * KP; i += 512) hbuf[i] = 0;

    float    creg[7][8];      // cell state, f32
    unsigned hstp[7][4];      // staged h_t, packed bf16 pairs (r even = lo half)
#pragma unroll
    for (int a = 0; a < 7; ++a)
#pragma unroll
        for (int b = 0; b < 8; ++b) creg[a][b] = 0.f;

    const float* WihL = (layer == 0) ? Wih0 : (layer == 1) ? Wih1 : Wih2;
    const int    wstr = (layer == 0) ? 3 : 403;
    const float* bxl  = bx + layer * 1600;
    const unsigned short* whhL  = whhP + (size_t)layer * WHH_EL;
    const unsigned short* wihhL = (layer > 0) ? wihhP + (size_t)(layer - 1) * WHH_EL : nullptr;
    unsigned short*       outL  = outs + (size_t)layer * OUT_EL;
    const unsigned short* prevL = (layer > 0) ? outs + (size_t)(layer - 1) * OUT_EL : nullptr;

    __syncthreads();

    for (int t = 0; t < T; ++t) {
        if (tid < B * 3)
            xbuf[tid] = X[(size_t)(tid / 3) * (T * 3) + t * 3 + (tid % 3)];
        if (layer > 0) {
            if ((tid & 127) == 0)
                __builtin_prefetch(prevL + (size_t)t * B * KP, 0, 1);   // global_prefetch_b8
            if (tid == 0) {
                while (__hip_atomic_load(&flags[(layer - 1) * T + t],
                                         __ATOMIC_ACQUIRE, __HIP_MEMORY_SCOPE_AGENT) == 0)
                    __builtin_amdgcn_s_sleep(2);
            }
        }
        __syncthreads();

#pragma unroll
        for (int gi = 0; gi < 7; ++gi) {
            const int g = wave + gi * 16;
            if (g < 100) {
                const int mt = g / 25, nt = g % 25;
                const int m  = mt * 16 + (lane & 15);
                v8f acc0 = {}, acc1 = {}, acc2 = {}, acc3 = {};

                // ---- h_{t-1} @ Whh^T : 13 k-steps x 4 gate tiles ----
                const unsigned short* hrow = hbuf + m * KP;
#pragma unroll 1
                for (int ks = 0; ks < KS; ++ks) {
                    AFrag a;
                    const int kb = ks * 32 + ((lane & 16) ? 8 : 0);
#pragma unroll
                    for (int vv = 0; vv < 8; ++vv) {
                        const int k = kb + ((vv & 4) ? 16 : 0) + ((vv & 3) << 1);
                        a.u[vv] = *(const unsigned int*)(hrow + k);      // -> ds_load_b128 x2
                    }
                    const v16bf* b0 = (const v16bf*)(whhL + ((((size_t)(nt)      * KS + ks) * 32 + lane) << 4));
                    const v16bf* b1 = (const v16bf*)(whhL + ((((size_t)(25 + nt) * KS + ks) * 32 + lane) << 4));
                    const v16bf* b2 = (const v16bf*)(whhL + ((((size_t)(50 + nt) * KS + ks) * 32 + lane) << 4));
                    const v16bf* b3 = (const v16bf*)(whhL + ((((size_t)(75 + nt) * KS + ks) * 32 + lane) << 4));
                    acc0 = __builtin_amdgcn_wmma_f32_16x16x32_bf16(false, a.v, false, *b0, (short)0, acc0, false, false);
                    acc1 = __builtin_amdgcn_wmma_f32_16x16x32_bf16(false, a.v, false, *b1, (short)0, acc1, false, false);
                    acc2 = __builtin_amdgcn_wmma_f32_16x16x32_bf16(false, a.v, false, *b2, (short)0, acc2, false, false);
                    acc3 = __builtin_amdgcn_wmma_f32_16x16x32_bf16(false, a.v, false, *b3, (short)0, acc3, false, false);
                }

                // ---- out_{L-1}[t] @ Wih[:,3:]^T (layers 1,2) ----
                if (layer > 0) {
                    const unsigned short* prow = prevL + ((size_t)t * B + m) * KP;
#pragma unroll 1
                    for (int ks = 0; ks < KS; ++ks) {
                        AFrag a;
                        const int kb = ks * 32 + ((lane & 16) ? 8 : 0);
#pragma unroll
                        for (int vv = 0; vv < 8; ++vv) {
                            const int k = kb + ((vv & 4) ? 16 : 0) + ((vv & 3) << 1);
                            a.u[vv] = *(const unsigned int*)(prow + k);   // -> global_load_b128 x2
                        }
                        const v16bf* b0 = (const v16bf*)(wihhL + ((((size_t)(nt)      * KS + ks) * 32 + lane) << 4));
                        const v16bf* b1 = (const v16bf*)(wihhL + ((((size_t)(25 + nt) * KS + ks) * 32 + lane) << 4));
                        const v16bf* b2 = (const v16bf*)(wihhL + ((((size_t)(50 + nt) * KS + ks) * 32 + lane) << 4));
                        const v16bf* b3 = (const v16bf*)(wihhL + ((((size_t)(75 + nt) * KS + ks) * 32 + lane) << 4));
                        acc0 = __builtin_amdgcn_wmma_f32_16x16x32_bf16(false, a.v, false, *b0, (short)0, acc0, false, false);
                        acc1 = __builtin_amdgcn_wmma_f32_16x16x32_bf16(false, a.v, false, *b1, (short)0, acc1, false, false);
                        acc2 = __builtin_amdgcn_wmma_f32_16x16x32_bf16(false, a.v, false, *b2, (short)0, acc2, false, false);
                        acc3 = __builtin_amdgcn_wmma_f32_16x16x32_bf16(false, a.v, false, *b3, (short)0, acc3, false, false);
                    }
                }

                // ---- gates: add bias + x-projection (K=3), activations ----
                const int hc = nt * 16 + (lane & 15);
                const float bx0 = bxl[hc],        bx1 = bxl[400 + hc];
                const float bx2 = bxl[800 + hc],  bx3 = bxl[1200 + hc];
                const float* wi = WihL + (size_t)hc * wstr;
                const float w00 = wi[0], w01 = wi[1], w02 = wi[2];
                const float* wf = WihL + (size_t)(400 + hc) * wstr;
                const float w10 = wf[0], w11 = wf[1], w12 = wf[2];
                const float* wg = WihL + (size_t)(800 + hc) * wstr;
                const float w20 = wg[0], w21 = wg[1], w22 = wg[2];
                const float* wo = WihL + (size_t)(1200 + hc) * wstr;
                const float w30 = wo[0], w31 = wo[1], w32 = wo[2];
                unsigned short hlo = 0;
#pragma unroll
                for (int r = 0; r < 8; ++r) {
                    const int mm = mt * 16 + r + ((lane & 16) ? 8 : 0);
                    const float x0 = xbuf[mm * 3 + 0];
                    const float x1 = xbuf[mm * 3 + 1];
                    const float x2 = xbuf[mm * 3 + 2];
                    const float pi = acc0[r] + bx0 + x0 * w00 + x1 * w01 + x2 * w02;
                    const float pf = acc1[r] + bx1 + x0 * w10 + x1 * w11 + x2 * w12;
                    const float pg = acc2[r] + bx2 + x0 * w20 + x1 * w21 + x2 * w22;
                    const float po = acc3[r] + bx3 + x0 * w30 + x1 * w31 + x2 * w32;
                    const float c  = sigf(pf) * creg[gi][r] + sigf(pi) * tanhf(pg);
                    creg[gi][r] = c;
                    const unsigned short hb = f2bf(sigf(po) * tanhf(c));
                    if ((r & 1) == 0) hlo = hb;
                    else              hstp[gi][r >> 1] = (unsigned)hlo | ((unsigned)hb << 16);
                }
            }
        }

        __syncthreads();   // all reads of hbuf for step t complete

#pragma unroll
        for (int gi = 0; gi < 7; ++gi) {
            const int g = wave + gi * 16;
            if (g < 100) {
                const int mt = g / 25, nt = g % 25;
                const int hc = nt * 16 + (lane & 15);
#pragma unroll
                for (int j = 0; j < 4; ++j) {
                    const unsigned pw = hstp[gi][j];
                    const unsigned short lo = (unsigned short)(pw & 0xffffu);
                    const unsigned short hi = (unsigned short)(pw >> 16);
                    const int mmlo = mt * 16 + 2 * j + ((lane & 16) ? 8 : 0);
                    hbuf[mmlo * KP + hc]       = lo;
                    hbuf[(mmlo + 1) * KP + hc] = hi;
                    outL[((size_t)t * B + mmlo) * KP + hc]     = lo;
                    outL[((size_t)t * B + mmlo + 1) * KP + hc] = hi;
                }
            }
        }
        for (int p = tid; p < B * 16; p += 512)   // zero K-pad columns for this t
            outL[((size_t)t * B + (p >> 4)) * KP + Hh + (p & 15)] = 0;

        __syncthreads();
        if (tid == 0) {
            __threadfence();
            __hip_atomic_store(&flags[layer * T + t], 1,
                               __ATOMIC_RELEASE, __HIP_MEMORY_SCOPE_AGENT);
        }
    }
}

// ---------------------------------------------------------------------------
// Final linear: per t, [64 x 1248(bf16)] @ [1248 x 128] -> [64 x 121] f32
// ---------------------------------------------------------------------------
__global__ void __launch_bounds__(256)
k_linear(const unsigned short* __restrict__ outs,
         const unsigned short* __restrict__ wlP,
         const float* __restrict__ bl, float* __restrict__ out) {
    const int t    = blockIdx.x;
    const int tid  = threadIdx.x;
    const int lane = tid & 31;
    const int nt   = tid >> 5;            // 8 waves = 8 n-tiles (OUT pad 128)
    for (int mt = 0; mt < 4; ++mt) {
        v8f acc = {};
#pragma unroll 1
        for (int ks = 0; ks < 39; ++ks) {
            const int seg = ks / KS, ksl = ks % KS;
            const unsigned short* prow =
                outs + (size_t)seg * OUT_EL + ((size_t)t * B + mt * 16 + (lane & 15)) * KP;
            AFrag a;
            const int kb = ksl * 32 + ((lane & 16) ? 8 : 0);
#pragma unroll
            for (int vv = 0; vv < 8; ++vv) {
                const int k = kb + ((vv & 4) ? 16 : 0) + ((vv & 3) << 1);
                a.u[vv] = *(const unsigned int*)(prow + k);
            }
            const v16bf* bfr = (const v16bf*)(wlP + (((size_t)(nt * 39 + ks) * 32 + lane) << 4));
            acc = __builtin_amdgcn_wmma_f32_16x16x32_bf16(false, a.v, false, *bfr, (short)0, acc, false, false);
        }
        const int o = nt * 16 + (lane & 15);
        if (o < 121) {
            const float bb = bl[o];
#pragma unroll
            for (int r = 0; r < 8; ++r) {
                const int b = mt * 16 + r + ((lane & 16) ? 8 : 0);
                out[((size_t)b * T + t) * 121 + o] = acc[r] + bb;
            }
        }
    }
}

// ---------------------------------------------------------------------------
extern "C" void kernel_launch(void* const* d_in, const int* in_sizes, int n_in,
                              void* d_out, int out_size, void* d_ws, size_t ws_size,
                              hipStream_t stream) {
    const float* X    = (const float*)d_in[0];
    const float* Wih0 = (const float*)d_in[1];
    const float* Whh0 = (const float*)d_in[2];
    const float* bih0 = (const float*)d_in[3];
    const float* bhh0 = (const float*)d_in[4];
    const float* Wih1 = (const float*)d_in[5];
    const float* Whh1 = (const float*)d_in[6];
    const float* bih1 = (const float*)d_in[7];
    const float* bhh1 = (const float*)d_in[8];
    const float* Wih2 = (const float*)d_in[9];
    const float* Whh2 = (const float*)d_in[10];
    const float* bih2 = (const float*)d_in[11];
    const float* bhh2 = (const float*)d_in[12];
    const float* Wl   = (const float*)d_in[13];
    const float* bl   = (const float*)d_in[14];

    char* ws = (char*)d_ws;
    int*            flags = (int*)(ws + OFF_FLAGS);
    float*          bx    = (float*)(ws + OFF_BX);
    unsigned short* whhP  = (unsigned short*)(ws + OFF_WHH);
    unsigned short* wihhP = (unsigned short*)(ws + OFF_WIHH);
    unsigned short* wlP   = (unsigned short*)(ws + OFF_WL);
    unsigned short* outs  = (unsigned short*)(ws + OFF_OUT);

    k_init_flags<<<(3 * T + 255) / 256, 256, 0, stream>>>(flags);
    k_pack_bias<<<(4800 + 255) / 256, 256, 0, stream>>>(bih0, bhh0, bih1, bhh1, bih2, bhh2, bx);

    const int wpn = (int)((WHH_EL + 255) / 256);
    k_pack_w<<<wpn, 256, 0, stream>>>(Whh0, whhP + 0 * WHH_EL, 400, 0);
    k_pack_w<<<wpn, 256, 0, stream>>>(Whh1, whhP + 1 * WHH_EL, 400, 0);
    k_pack_w<<<wpn, 256, 0, stream>>>(Whh2, whhP + 2 * WHH_EL, 400, 0);
    k_pack_w<<<wpn, 256, 0, stream>>>(Wih1, wihhP + 0 * WHH_EL, 403, 3);
    k_pack_w<<<wpn, 256, 0, stream>>>(Wih2, wihhP + 1 * WHH_EL, 403, 3);
    k_pack_wl<<<(int)((WL_EL + 255) / 256), 256, 0, stream>>>(Wl, wlP);

    k_lstm<<<3, 512, 0, stream>>>(X, Wih0, Wih1, Wih2, bx, whhP, wihhP, outs, flags);
    k_linear<<<T, 256, 0, stream>>>(outs, wlP, bl, (float*)d_out);
}
